// CausalSelfAttention_30700426232399
// MI455X (gfx1250) — matrix operand order repaired and checked by reading, not compiled, not run on hardware
//
#include <hip/hip_runtime.h>
#include <hip/hip_bf16.h>

// ---- problem constants (B=2, T=2048, D=1024, H=16, HD=64) ----
#define BB 2
#define TT 2048
#define DD 1024
#define HH 16
#define HD 64
#define MROWS (BB * TT)   // 4096

typedef __bf16 bf16;
typedef __attribute__((ext_vector_type(16))) __bf16 v16bf;
typedef __attribute__((ext_vector_type(8)))  float  v8f;

#define WMMA_BF16(a, b, c) \
  __builtin_amdgcn_wmma_f32_16x16x32_bf16(false, (a), false, (b), (short)0, (c), false, false)

// ---- A-matrix tile load (16x32 bf16, ISA 7.12.2 interleaved layout) ----
// per lane: row = lane&15; chunks of 8 at k = (lane>=16 ? 8 : 0) and +16
__device__ __forceinline__ v16bf load_a16(const bf16* tile, int ld, int lane) {
  int row = lane & 15;
  int kb  = (lane >> 4) << 3;             // 0 or 8
  const bf16* p = tile + (size_t)row * ld + kb;
  union { v16bf v; uint4 q[2]; } u;
  u.q[0] = *reinterpret_cast<const uint4*>(p);
  u.q[1] = *reinterpret_cast<const uint4*>(p + 16);
  return u.v;
}

// ---- B-matrix tile load (32x16 bf16). Caller supplies B as 16 rows (N) of
// contiguous K: element (k,n) at tile[n*ld + k]. Per lane: col = lane&15,
// chunks of 8 at k = (lane>=16 ? 8 : 0) and +16. ----
__device__ __forceinline__ v16bf load_b16(const bf16* tile, int ld, int lane) {
  int col = lane & 15;
  int kb  = (lane >> 4) << 3;             // 0 or 8
  const bf16* p = tile + (size_t)col * ld + kb;
  union { v16bf v; uint4 q[2]; } u;
  u.q[0] = *reinterpret_cast<const uint4*>(p);
  u.q[1] = *reinterpret_cast<const uint4*>(p + 16);
  return u.v;
}

// ---- f32 -> bf16 convert ----
__global__ void __launch_bounds__(256)
cvt_f32_bf16(const float* __restrict__ in, bf16* __restrict__ out, int n) {
  int i = (blockIdx.x * blockDim.x + threadIdx.x) * 4;
  if (i + 3 < n) {
    float4 f = *reinterpret_cast<const float4*>(in + i);
    union { bf16 b[4]; uint2 u; } o;
    o.b[0] = (bf16)f.x; o.b[1] = (bf16)f.y; o.b[2] = (bf16)f.z; o.b[3] = (bf16)f.w;
    *reinterpret_cast<uint2*>(out + i) = o.u;
  }
}

// ---- GEMM: C[M,N] = A[M,K] * B[N,K]^T + bias.
// Wave tile 32M x 64N (8 WMMAs / k-chunk, 12 loads / k-chunk). Software
// pipeline with explicit ping-pong stages (no register rotation copies):
// while computing stage X, the other stage's loads are in flight.
// Block = 8 waves (2M x 4N) -> block tile 64M x 256N. bf16 ops, f32 acc.
// mode 0: store bf16 as [b,h,t,d];  mode 2: store bf16 as [b,h,d,t] (V^T);
// mode 3: store f32 row-major to outf.  Requires K % 64 == 0. ----
__global__ void __launch_bounds__(256)
gemm_bf16_wmma(const bf16* __restrict__ A, const bf16* __restrict__ Bm,
               const float* __restrict__ bias,
               bf16* __restrict__ outb, float* __restrict__ outf,
               int K, int N, int mode) {
  int lane = threadIdx.x & 31;
  int w    = threadIdx.x >> 5;
  int m0 = blockIdx.x * 64  + (w & 1) * 32;
  int n0 = blockIdx.y * 256 + (w >> 1) * 64;
  int hi = lane >> 4;
  int nn = lane & 15;

  const bf16* Arow0 = A + (size_t)m0 * K;
  const bf16* Arow1 = A + (size_t)(m0 + 16) * K;
  const bf16* Brow[4];
#pragma unroll
  for (int nt = 0; nt < 4; ++nt) Brow[nt] = Bm + (size_t)(n0 + nt * 16) * K;

  v8f accL[4] = {};
  v8f accH[4] = {};

  // stage X <- chunk 0
  v16bf a0x = load_a16(Arow0, K, lane);
  v16bf a1x = load_a16(Arow1, K, lane);
  v16bf bx[4];
#pragma unroll
  for (int nt = 0; nt < 4; ++nt) bx[nt] = load_b16(Brow[nt], K, lane);

  v16bf a0y, a1y, by[4];

  int k0 = 0;
  for (; k0 + 64 < K; k0 += 64) {
    // load stage Y <- k0+32
    a0y = load_a16(Arow0 + k0 + 32, K, lane);
    a1y = load_a16(Arow1 + k0 + 32, K, lane);
#pragma unroll
    for (int nt = 0; nt < 4; ++nt) by[nt] = load_b16(Brow[nt] + k0 + 32, K, lane);
    // compute stage X (chunk k0)
#pragma unroll
    for (int nt = 0; nt < 4; ++nt) {
      accL[nt] = WMMA_BF16(a0x, bx[nt], accL[nt]);
      accH[nt] = WMMA_BF16(a1x, bx[nt], accH[nt]);
    }
    // load stage X <- k0+64
    a0x = load_a16(Arow0 + k0 + 64, K, lane);
    a1x = load_a16(Arow1 + k0 + 64, K, lane);
#pragma unroll
    for (int nt = 0; nt < 4; ++nt) bx[nt] = load_b16(Brow[nt] + k0 + 64, K, lane);
    // compute stage Y (chunk k0+32)
#pragma unroll
    for (int nt = 0; nt < 4; ++nt) {
      accL[nt] = WMMA_BF16(a0y, by[nt], accL[nt]);
      accH[nt] = WMMA_BF16(a1y, by[nt], accH[nt]);
    }
  }
  // tail: chunks k0 (staged in X) and k0+32
  a0y = load_a16(Arow0 + k0 + 32, K, lane);
  a1y = load_a16(Arow1 + k0 + 32, K, lane);
#pragma unroll
  for (int nt = 0; nt < 4; ++nt) by[nt] = load_b16(Brow[nt] + k0 + 32, K, lane);
#pragma unroll
  for (int nt = 0; nt < 4; ++nt) {
    accL[nt] = WMMA_BF16(a0x, bx[nt], accL[nt]);
    accH[nt] = WMMA_BF16(a1x, bx[nt], accH[nt]);
  }
#pragma unroll
  for (int nt = 0; nt < 4; ++nt) {
    accL[nt] = WMMA_BF16(a0y, by[nt], accL[nt]);
    accH[nt] = WMMA_BF16(a1y, by[nt], accH[nt]);
  }

  // store
#pragma unroll
  for (int nt = 0; nt < 4; ++nt) {
    int n = n0 + nt * 16 + nn;
    float bval = bias[n];
    int hh = n >> 6, dd = n & 63;      // head / dim for modes 0,2
#pragma unroll
    for (int half = 0; half < 2; ++half) {
      const v8f& av = half ? accH[nt] : accL[nt];
#pragma unroll
      for (int r = 0; r < 8; ++r) {
        int mrow = m0 + half * 16 + r + hi * 8;
        float val = av[r] + bval;
        if (mode == 3) {
          outf[(size_t)mrow * N + n] = val;
        } else {
          int bb = mrow >> 11;         // / TT
          int tt = mrow & (TT - 1);
          size_t idx = (mode == 0)
            ? ((size_t)(bb * HH + hh) * TT + tt) * HD + dd    // [b,h,t,d]
            : ((size_t)(bb * HH + hh) * HD + dd) * TT + tt;   // [b,h,d,t]
          outb[idx] = (bf16)val;
        }
      }
    }
  }
}

// ---- Flash attention: one wave per 16-row Q tile per (b,h). 4 waves/block.
// Online softmax over 32-key chunks. V tiles issued before the softmax VALU
// work; next chunk's K tiles prefetched (address-clamped, unconditional)
// before the P*V WMMAs. ----
__global__ void __launch_bounds__(128)
flash_attn_wmma(const bf16* __restrict__ Qb, const bf16* __restrict__ Kb,
                const bf16* __restrict__ Vt, bf16* __restrict__ Y) {
  __shared__ __align__(16) bf16 sP[4][16][32];   // per-wave P staging (4 KB)

  int lane = threadIdx.x & 31;
  int wv   = threadIdx.x >> 5;
  int gw   = blockIdx.x * 4 + wv;
  int qt   = gw & (TT / 16 - 1);       // 0..127
  int bh   = gw >> 7;                  // 0..31
  int q0   = qt << 4;
  int hi   = lane >> 4;
  int nn   = lane & 15;

  const bf16* Qbase = Qb + ((size_t)bh * TT + q0) * HD;
  const bf16* Kbase = Kb + (size_t)bh * TT * HD;
  const bf16* Vbase = Vt + (size_t)bh * HD * TT;

  v16bf qa0 = load_a16(Qbase + 0,  HD, lane);
  v16bf qa1 = load_a16(Qbase + 32, HD, lane);

  v8f acc0 = {}, acc1 = {}, acc2 = {}, acc3 = {};
  float m_run[8], l_run[8];
#pragma unroll
  for (int r = 0; r < 8; ++r) { m_run[r] = -1e30f; l_run[r] = 0.0f; }

  int kend = q0 + 16;                  // need keys [0, q0+15]

  // stage K tiles for chunk 0
  v16bf kb0a = load_b16(Kbase + 0,                    HD, lane);
  v16bf kb0b = load_b16(Kbase + 32,                   HD, lane);
  v16bf kb1a = load_b16(Kbase + (size_t)16 * HD,      HD, lane);
  v16bf kb1b = load_b16(Kbase + (size_t)16 * HD + 32, HD, lane);

  for (int k0 = 0; k0 < kend; k0 += 32) {
    // issue V tiles for this chunk early (independent of S / softmax)
    v16bf vb0 = load_b16(Vbase + (size_t)(0)  * TT + k0, TT, lane);
    v16bf vb1 = load_b16(Vbase + (size_t)(16) * TT + k0, TT, lane);
    v16bf vb2 = load_b16(Vbase + (size_t)(32) * TT + k0, TT, lane);
    v16bf vb3 = load_b16(Vbase + (size_t)(48) * TT + k0, TT, lane);

    // ---- S = Q K^T for 32 keys (two 16-key tiles) ----
    v8f s0 = {}, s1 = {};
    s0 = WMMA_BF16(qa0, kb0a, s0);
    s0 = WMMA_BF16(qa1, kb0b, s0);
    s1 = WMMA_BF16(qa0, kb1a, s1);
    s1 = WMMA_BF16(qa1, kb1b, s1);

    // prefetch next chunk's K tiles (clamped address: last iteration
    // harmlessly reloads the current chunk; no conditional codegen)
    int kn = k0 + 32;
    int kp = (kn < kend) ? kn : k0;
    v16bf kb0an = load_b16(Kbase + (size_t)(kp)      * HD + 0,  HD, lane);
    v16bf kb0bn = load_b16(Kbase + (size_t)(kp)      * HD + 32, HD, lane);
    v16bf kb1an = load_b16(Kbase + (size_t)(kp + 16) * HD + 0,  HD, lane);
    v16bf kb1bn = load_b16(Kbase + (size_t)(kp + 16) * HD + 32, HD, lane);

    bool domask = (k0 + 31 > q0);
#pragma unroll
    for (int r = 0; r < 8; ++r) {
      int m = r + hi * 8;              // query row within tile (C/D layout)
      float v0 = s0[r] * 0.125f;       // 1/sqrt(64)
      float v1 = s1[r] * 0.125f;
      if (domask) {
        if (k0 + nn      > q0 + m) v0 = -1e30f;
        if (k0 + 16 + nn > q0 + m) v1 = -1e30f;
      }
      float rmax = fmaxf(v0, v1);
#pragma unroll
      for (int off = 1; off < 16; off <<= 1)
        rmax = fmaxf(rmax, __shfl_xor(rmax, off, 32));
      float mn    = fmaxf(m_run[r], rmax);
      float alpha = __expf(m_run[r] - mn);
      float p0    = __expf(v0 - mn);
      float p1    = __expf(v1 - mn);
      float rs    = p0 + p1;
#pragma unroll
      for (int off = 1; off < 16; off <<= 1)
        rs += __shfl_xor(rs, off, 32);
      l_run[r] = l_run[r] * alpha + rs;
      m_run[r] = mn;
      acc0[r] *= alpha; acc1[r] *= alpha; acc2[r] *= alpha; acc3[r] *= alpha;
      sP[wv][m][nn]      = (bf16)p0;   // stage P for layout change
      sP[wv][m][nn + 16] = (bf16)p1;
    }
    asm volatile("s_wait_dscnt 0x0" ::: "memory");

    // ---- reload P in A-matrix layout (row = nn, interleaved k chunks) ----
    v16bf pa;
    {
      const bf16* p = &sP[wv][nn][hi * 8];
      union { v16bf v; uint4 q[2]; } u;
      u.q[0] = *reinterpret_cast<const uint4*>(p);
      u.q[1] = *reinterpret_cast<const uint4*>(p + 16);
      pa = u.v;
    }

    // ---- O += P V  (V stored transposed [d, t], contiguous along t) ----
    acc0 = WMMA_BF16(pa, vb0, acc0);
    acc1 = WMMA_BF16(pa, vb1, acc1);
    acc2 = WMMA_BF16(pa, vb2, acc2);
    acc3 = WMMA_BF16(pa, vb3, acc3);

    kb0a = kb0an; kb0b = kb0bn; kb1a = kb1an; kb1b = kb1bn;
  }

  // ---- normalize and store y[b, t, h*64 + d] as bf16 ----
  int b = bh >> 4;
  int h = bh & 15;
#pragma unroll
  for (int r = 0; r < 8; ++r) {
    int m = r + hi * 8;
    float inv = 1.0f / l_run[r];
    size_t base = ((size_t)(b * TT + q0 + m)) * DD + h * HD + nn;
    Y[base +  0] = (bf16)(acc0[r] * inv);
    Y[base + 16] = (bf16)(acc1[r] * inv);
    Y[base + 32] = (bf16)(acc2[r] * inv);
    Y[base + 48] = (bf16)(acc3[r] * inv);
  }
}

extern "C" void kernel_launch(void* const* d_in, const int* in_sizes, int n_in,
                              void* d_out, int out_size, void* d_ws, size_t ws_size,
                              hipStream_t stream) {
  const float* x   = (const float*)d_in[0];
  const float* Wq  = (const float*)d_in[1];
  const float* bq  = (const float*)d_in[2];
  const float* Wk  = (const float*)d_in[3];
  const float* bk  = (const float*)d_in[4];
  const float* Wv  = (const float*)d_in[5];
  const float* bvv = (const float*)d_in[6];
  const float* Wp  = (const float*)d_in[7];
  const float* bp  = (const float*)d_in[8];
  float* out = (float*)d_out;

  // workspace layout (48 MiB total)
  char* ws = (char*)d_ws;
  bf16* xb  = (bf16*)(ws);                         // 8 MiB: x bf16 [4096,1024]
  bf16* wqb = (bf16*)(ws + (size_t)( 8 << 20));    // 2 MiB each
  bf16* wkb = (bf16*)(ws + (size_t)(10 << 20));
  bf16* wvb = (bf16*)(ws + (size_t)(12 << 20));
  bf16* wpb = (bf16*)(ws + (size_t)(14 << 20));
  bf16* Qb  = (bf16*)(ws + (size_t)(16 << 20));    // 8 MiB [b,h,t,d]
  bf16* Kb  = (bf16*)(ws + (size_t)(24 << 20));    // 8 MiB [b,h,t,d]
  bf16* Vt  = (bf16*)(ws + (size_t)(32 << 20));    // 8 MiB [b,h,d,t]
  bf16* Yb  = (bf16*)(ws + (size_t)(40 << 20));    // 8 MiB [4096,1024]

  const int nx = MROWS * DD;     // 4M
  const int nw = DD * DD;        // 1M
  cvt_f32_bf16<<<nx / 1024, 256, 0, stream>>>(x,  xb,  nx);
  cvt_f32_bf16<<<nw / 1024, 256, 0, stream>>>(Wq, wqb, nw);
  cvt_f32_bf16<<<nw / 1024, 256, 0, stream>>>(Wk, wkb, nw);
  cvt_f32_bf16<<<nw / 1024, 256, 0, stream>>>(Wv, wvb, nw);
  cvt_f32_bf16<<<nw / 1024, 256, 0, stream>>>(Wp, wpb, nw);

  dim3 g(MROWS / 64, DD / 256);  // (64, 4)
  gemm_bf16_wmma<<<g, 256, 0, stream>>>(xb, wqb, bq,  Qb, nullptr, DD, DD, 0);
  gemm_bf16_wmma<<<g, 256, 0, stream>>>(xb, wkb, bk,  Kb, nullptr, DD, DD, 0);
  gemm_bf16_wmma<<<g, 256, 0, stream>>>(xb, wvb, bvv, Vt, nullptr, DD, DD, 2);

  // 4096 waves total: one per (b,h,q-tile); 4 waves per block
  flash_attn_wmma<<<(BB * HH * (TT / 16)) / 4, 128, 0, stream>>>(Qb, Kb, Vt, Yb);

  gemm_bf16_wmma<<<g, 256, 0, stream>>>(Yb, wpb, bp, nullptr, out, DD, DD, 3);
}
